// GCNClassifier_72275709657222
// MI455X (gfx1250) — matrix operand order repaired and checked by reading, not compiled, not run on hardware
//
#include <hip/hip_runtime.h>
#include <hip/hip_bf16.h>

#define IN_DIM  512
#define HID_DIM 256
#define OUT_DIM 64

typedef __attribute__((ext_vector_type(16))) __bf16 v16bf;
typedef __attribute__((ext_vector_type(8)))  float  v8f;
typedef int v4i __attribute__((vector_size(16)));

union Frag   { uint4 q[2]; v16bf v; };
union PackBF { __bf16 h[4]; uint2 u; };
union Pack2  { __bf16 h[2]; unsigned u; };

#if defined(__has_builtin)
#if __has_builtin(__builtin_amdgcn_global_load_async_to_lds_b128)
#define HAVE_ASYNC_LDS 1
#endif
#endif

#ifdef HAVE_ASYNC_LDS
__device__ __forceinline__ void wait_async0() {
#if __has_builtin(__builtin_amdgcn_s_wait_asynccnt)
  __builtin_amdgcn_s_wait_asynccnt(0);
#else
  asm volatile("s_wait_asynccnt 0x0" ::: "memory");
#endif
}
#endif

// ---------------- degree / normalization ----------------
__global__ void k_deg_init(float* __restrict__ deg, int N) {
  int i = blockIdx.x * blockDim.x + threadIdx.x;
  if (i < N) deg[i] = 1.0f;               // self-loop contributes 1
}

__global__ void k_deg_count(const long long* __restrict__ dstI, float* __restrict__ deg, int E) {
  int i = blockIdx.x * blockDim.x + threadIdx.x;
  if (i < E) unsafeAtomicAdd(&deg[(int)dstI[i]], 1.0f);
}

__global__ void k_dis(float* __restrict__ deg, int N) {
  int i = blockIdx.x * blockDim.x + threadIdx.x;
  if (i < N) deg[i] = rsqrtf(deg[i]);     // deg >= 1 due to self-loop
}

// ---------------- bf16 WMMA GEMM:  out = scale(m) * (A @ B) written to two buffers ----------------
// A: [M,K] (f32 or bf16), B: [K,Nn] f32 row-major. Block tile 128 x NB, 8 waves (4x2),
// wave tile 32 x (NB/2), NFRAG = NB/32 accumulator columns per wave.
template<int NB, bool A_BF16, bool HAS_DIS>
__global__ __launch_bounds__(256)
void gcn_gemm(const void* __restrict__ Aptr, const float* __restrict__ Bw,
              const float* __restrict__ dis,
              float* __restrict__ out1, float* __restrict__ out2,
              int M, int K, int Nn)
{
  constexpr int NFRAG = NB / 32;     // 16-wide B fragments per wave
  constexpr int WT    = NB / 2;      // wave tile width

  __shared__ __attribute__((aligned(16))) __bf16 lA[128][40];  // [row][k]  pitch 80B
  __shared__ __attribute__((aligned(16))) __bf16 lB[NB][40];   // [n][k]    transposed tile

  const int tid  = threadIdx.x;
  const int lane = tid & 31;
  const int wave = tid >> 5;
  const int wm   = wave >> 1;        // 0..3  (M direction)
  const int wn   = wave & 1;         // 0..1  (N direction)
  const int half = lane >> 4;        // 0/1
  const int l15  = lane & 15;
  const int m0   = blockIdx.x * 128;
  const int n0   = blockIdx.y * NB;

  const float*  Af = (const float*)Aptr;
  const __bf16* Ab = (const __bf16*)Aptr;

  v8f acc[2][NFRAG];
#pragma unroll
  for (int s = 0; s < 2; ++s)
#pragma unroll
    for (int t = 0; t < NFRAG; ++t) acc[s][t] = (v8f){};

  for (int k0 = 0; k0 < K; k0 += 32) {
    const int kpf = (k0 + 32 < K) ? 32 : 0;    // branchless prefetch offset
    // ---- stage A tile 128x32 into LDS as bf16 ----
    if (A_BF16) {
#ifdef HAVE_ASYNC_LDS
      // async global->LDS DMA (16B chunks, 512 total, 2 per thread), tracked by ASYNCcnt
#pragma unroll
      for (int p = 0; p < 2; ++p) {
        int chunk = p * 256 + tid;
        int row   = chunk >> 2;            // 0..127
        int c     = (chunk & 3) * 8;       // 0,8,16,24
        int gr    = m0 + row; if (gr >= M) gr = M - 1;
        const __bf16* gp = Ab + (size_t)gr * K + k0 + c;
        __builtin_prefetch(gp + kpf, 0, 3);
        __builtin_amdgcn_global_load_async_to_lds_b128(
            (__attribute__((address_space(1))) v4i*)gp,
            (__attribute__((address_space(3))) v4i*)&lA[row][c], 0, 0);
      }
      wait_async0();
#else
      int r = tid >> 3;                    // 0..31
      int c = (tid & 7) * 4;               // 0..28
#pragma unroll
      for (int p = 0; p < 4; ++p) {
        int row = p * 32 + r;
        int gr  = m0 + row; if (gr >= M) gr = M - 1;
        const __bf16* src = Ab + (size_t)gr * K + k0 + c;
        __builtin_prefetch(src + kpf, 0, 3);
        *(uint2*)&lA[row][c] = *(const uint2*)src;
      }
#endif
    } else {
      int r = tid >> 3;                    // 0..31
      int c = (tid & 7) * 4;               // 0..28
#pragma unroll
      for (int p = 0; p < 4; ++p) {
        int row = p * 32 + r;
        int gr  = m0 + row; if (gr >= M) gr = M - 1;
        const float* src = Af + (size_t)gr * K + k0 + c;
        __builtin_prefetch(src + kpf, 0, 3);
        float4 v = *(const float4*)src;
        PackBF pk;
        pk.h[0] = (__bf16)v.x; pk.h[1] = (__bf16)v.y;
        pk.h[2] = (__bf16)v.z; pk.h[3] = (__bf16)v.w;
        *(uint2*)&lA[row][c] = pk.u;       // single 8B DS store
      }
    }
    // ---- stage B tile 32 x NB, transposed into lB[n][k] ----
    // each thread: 2 consecutive k rows x (NB/16) consecutive n cols
    {
      constexpr int NU = NB / 16;
      int kp = (tid >> 4) * 2;             // 0,2,...,30
      int nq = (tid & 15) * NU;
      const float* s0 = Bw + (size_t)(k0 + kp) * Nn + n0 + nq;
      const float* s1 = s0 + Nn;
#pragma unroll
      for (int u = 0; u < NU; ++u) {
        Pack2 pk;
        pk.h[0] = (__bf16)s0[u];           // k = kp
        pk.h[1] = (__bf16)s1[u];           // k = kp+1  -> one v_cvt_pk_bf16_f32
        *(unsigned*)&lB[nq + u][kp] = pk.u;   // single 4B DS store
      }
    }
    __syncthreads();

    // ---- fragments per ISA 16-bit A/B layouts (two ds_load_b128 each) ----
    Frag a[2], b[NFRAG];
#pragma unroll
    for (int s = 0; s < 2; ++s) {
      const __bf16* p = &lA[wm*32 + s*16 + l15][half*8];
      a[s].q[0] = *(const uint4*)p;
      a[s].q[1] = *(const uint4*)(p + 16);
    }
#pragma unroll
    for (int t = 0; t < NFRAG; ++t) {
      const __bf16* p = &lB[wn*WT + t*16 + l15][half*8];
      b[t].q[0] = *(const uint4*)p;
      b[t].q[1] = *(const uint4*)(p + 16);
    }

#pragma unroll
    for (int s = 0; s < 2; ++s)
#pragma unroll
      for (int t = 0; t < NFRAG; ++t)
        acc[s][t] = __builtin_amdgcn_wmma_f32_16x16x32_bf16(
            false, a[s].v, false, b[t].v, (short)0, acc[s][t], false, false);
    __syncthreads();
  }

  // ---- epilogue: val = dis[m] * acc, written to both out buffers ----
  const bool full = (m0 + 128 <= M);       // block-uniform fast path
  const int  nb   = n0 + wn*WT + l15;      // column for t=0 (t adds +16)
#pragma unroll
  for (int s = 0; s < 2; ++s) {
    const int mb  = m0 + wm*32 + s*16 + half*8;   // C/D layout: VGPR r -> M = r + 8*half
    size_t base = (size_t)mb * Nn + nb;
#pragma unroll
    for (int r = 0; r < 8; ++r) {
      if (full || (mb + r) < M) {
        float sc = HAS_DIS ? dis[mb + r] : 1.0f;
        size_t o = base + (size_t)r * Nn;
#pragma unroll
        for (int t = 0; t < NFRAG; ++t) {
          float v = acc[s][t][r] * sc;
          out1[o + 16*t] = v;
          out2[o + 16*t] = v;
        }
      }
    }
  }
}

// ---------------- edge scatter-add: acc[dst] += feat[src], one edge per block ----------------
// blockDim == F, so srcI/dstI loads are block-uniform -> scalar loads.
template<int F>
__global__ void gcn_scatter(const float* __restrict__ feat, float* __restrict__ acc,
                            const long long* __restrict__ srcI, const long long* __restrict__ dstI)
{
  int e = blockIdx.x;
  int j = threadIdx.x;
  int s = (int)srcI[e];
  int d = (int)dstI[e];
  unsafeAtomicAdd(acc + (size_t)d * F + j, feat[(size_t)s * F + j]);
}

// ---------------- layer-1 epilogue: h2s = dis * relu(dis*acc1 + b1)  (bf16 for GEMM2) ----------------
__global__ void k_epi1(const float* __restrict__ acc1, const float* __restrict__ b1,
                       const float* __restrict__ dis, __bf16* __restrict__ h2s, int N)
{
  int idx = blockIdx.x * blockDim.x + threadIdx.x;
  if (idx >= N * HID_DIM) return;
  int n = idx >> 8, j = idx & (HID_DIM - 1);
  float dn = dis[n];
  float v  = fmaxf(dn * acc1[idx] + b1[j], 0.0f);
  h2s[idx] = (__bf16)(dn * v);
}

// ---------------- layer-2 epilogue (in place on d_out): out = dis*out + b2 ----------------
__global__ void k_epi2(float* __restrict__ out, const float* __restrict__ b2,
                       const float* __restrict__ dis, int N)
{
  int idx = blockIdx.x * blockDim.x + threadIdx.x;
  if (idx >= N * OUT_DIM) return;
  int n = idx >> 6, j = idx & (OUT_DIM - 1);
  out[idx] = dis[n] * out[idx] + b2[j];
}

extern "C" void kernel_launch(void* const* d_in, const int* in_sizes, int n_in,
                              void* d_out, int out_size, void* d_ws, size_t ws_size,
                              hipStream_t stream)
{
  (void)n_in; (void)out_size; (void)ws_size;
  const float*     x    = (const float*)d_in[0];
  const float*     W1   = (const float*)d_in[1];
  const float*     b1   = (const float*)d_in[2];
  const float*     W2   = (const float*)d_in[3];
  const float*     b2   = (const float*)d_in[4];
  const long long* eidx = (const long long*)d_in[5];   // int64 [2,E]
  const int N = in_sizes[0] / IN_DIM;
  const int E = in_sizes[5] / 2;
  const long long* srcI = eidx;
  const long long* dstI = eidx + E;
  float* out = (float*)d_out;

  // workspace carve-out
  char*  ws  = (char*)d_ws;
  size_t off = 0;
  auto alloc = [&](size_t bytes) {
    void* p = ws + off;
    off = (off + bytes + 255) & ~(size_t)255;
    return p;
  };
  float*  dis  = (float*)alloc((size_t)N * 4);                  // deg -> rsqrt in place
  float*  hs   = (float*)alloc((size_t)N * HID_DIM * 4);        // dis ⊙ (x@W1)
  float*  acc1 = (float*)alloc((size_t)N * HID_DIM * 4);        // init=hs (self loop), += scatter
  __bf16* h2s  = (__bf16*)alloc((size_t)N * HID_DIM * 2);       // layer-2 GEMM input
  float*  g    = (float*)alloc((size_t)N * OUT_DIM * 4);        // h2s@W2 (d_out is the accumulator)

  const int T = 256;
  k_deg_init <<<(N + T - 1) / T, T, 0, stream>>>(dis, N);
  k_deg_count<<<(E + T - 1) / T, T, 0, stream>>>(dstI, dis, E);
  k_dis      <<<(N + T - 1) / T, T, 0, stream>>>(dis, N);

  // Layer 1: hs = dis ⊙ (x @ W1); acc1 initialized to hs (self-loop folded)
  dim3 g1((N + 127) / 128, HID_DIM / 128);
  gcn_gemm<128, false, true><<<g1, 256, 0, stream>>>(x, W1, dis, hs, acc1, N, IN_DIM, HID_DIM);

  gcn_scatter<HID_DIM><<<E, HID_DIM, 0, stream>>>(hs, acc1, srcI, dstI);

  k_epi1<<<(N * HID_DIM + T - 1) / T, T, 0, stream>>>(acc1, b1, dis, h2s, N);

  // Layer 2: g = h2s @ W2  (dis already folded into h2s); d_out initialized to g
  dim3 g2((N + 127) / 128, OUT_DIM / 64);
  gcn_gemm<64, true, false><<<g2, 256, 0, stream>>>(h2s, W2, nullptr, g, out, N, HID_DIM, OUT_DIM);

  gcn_scatter<OUT_DIM><<<E, OUT_DIM, 0, stream>>>(g, out, srcI, dstI);

  k_epi2<<<(N * OUT_DIM + T - 1) / T, T, 0, stream>>>(out, b2, dis, N);
}
